// ReferenceGroupedMLP_55216099557791
// MI455X (gfx1250) — matrix operand order repaired and checked by reading, not compiled, not run on hardware
//
#include <hip/hip_runtime.h>

// Grouped SwiGLU MLP for MI455X (gfx1250), fused single kernel, v3.
//   gate = x @ gate_w ; up = x @ up_w ; h = silu(gate)*up ; out = h @ down_w
// v3: fragment loads batched ahead of the WMMA group so ds_load_b128s overlap
// matrix-pipe execution (graduated s_wait_dscnt instead of 0 before each
// WMMA). Otherwise: float4 staging loads, packed f16 LDS stores, double-
// buffered software pipeline (1 barrier / K-step), hidden tile in LDS.

typedef _Float16 f16;
typedef _Float16 f16x4 __attribute__((ext_vector_type(4)));
typedef _Float16 f16x8 __attribute__((ext_vector_type(8)));
typedef _Float16 v16h  __attribute__((ext_vector_type(16)));
typedef float    v8f   __attribute__((ext_vector_type(8)));
typedef float    f32x4 __attribute__((ext_vector_type(4)));

#define N_TOK 2048
#define NGRP  32
#define D_IN  1024
#define D_HID 2048
#define D_OUT 1024

#define BM  64            // rows per block
#define BN  64            // output cols per chunk
#define BK  32            // K step (matches 16x16x32 WMMA K)
#define LDK 40            // padded K stride of staged tiles (f16; 80 B rows)
#define LDH 2056          // padded hidden stride (f16; 4112 B rows)
#define SBUF (3 * BM * LDK)   // one stage buffer: xs + wg + wu (7680 f16)

// ---- WMMA fragment helpers (layouts per CDNA5 ISA 7.12.2, wave32) ----------
// A (16x32 f16): lane holds row M=lane%16; K = half*8+0..7 then 16+half*8+0..7.
__device__ __forceinline__ v16h frag_a(const f16* rowk, int half) {
  f16x8 lo = *(const f16x8*)(rowk + half * 8);
  f16x8 hi = *(const f16x8*)(rowk + 16 + half * 8);
  v16h r;
#pragma unroll
  for (int i = 0; i < 8; ++i) { r[i] = lo[i]; r[8 + i] = hi[i]; }
  return r;
}
// B (32x16 f16): lane holds col N=lane%16; K = half*8+0..7 then 16+half*8+0..7
// (tile staged transposed as [N][K]).
__device__ __forceinline__ v16h frag_b(const f16* nrow, int half) {
  f16x8 lo = *(const f16x8*)(nrow + half * 8);
  f16x8 hi = *(const f16x8*)(nrow + 16 + half * 8);
  v16h r;
#pragma unroll
  for (int i = 0; i < 8; ++i) { r[i] = lo[i]; r[8 + i] = hi[i]; }
  return r;
}
__device__ __forceinline__ v8f wmma16(v16h a, v16h b, v8f c) {
  return __builtin_amdgcn_wmma_f32_16x16x32_f16(false, a, false, b, (short)0, c,
                                                false, false);
}
__device__ __forceinline__ float silu(float v) { return v / (1.0f + __expf(-v)); }
__device__ __forceinline__ f16x4 cvt4(f32x4 v) {
  f16x4 r;
#pragma unroll
  for (int i = 0; i < 4; ++i) r[i] = (f16)v[i];
  return r;
}

// Convert + commit one staged tile set for phase A (x row-major, W transposed).
__device__ __forceinline__ void store_stageA(f16* xs, f16* wg, f16* wu,
                                             int mx, int qk, int kw, int n4,
                                             f32x4 rx0, f32x4 rx1, f32x4 rg0,
                                             f32x4 rg1, f32x4 ru0, f32x4 ru1) {
  *(f16x4*)(xs + mx * LDK + qk)        = cvt4(rx0);
  *(f16x4*)(xs + (mx + 32) * LDK + qk) = cvt4(rx1);
  f16x4 g0 = cvt4(rg0), g1 = cvt4(rg1), u0 = cvt4(ru0), u1 = cvt4(ru1);
#pragma unroll
  for (int j = 0; j < 4; ++j) {
    wg[(n4 + j) * LDK + kw]      = g0[j];
    wg[(n4 + j) * LDK + kw + 16] = g1[j];
    wu[(n4 + j) * LDK + kw]      = u0[j];
    wu[(n4 + j) * LDK + kw + 16] = u1[j];
  }
}

__global__ __launch_bounds__(256)
void grouped_swiglu_mlp(const float* __restrict__ x,
                        const float* __restrict__ gate_w,
                        const float* __restrict__ up_w,
                        const float* __restrict__ down_w,
                        float* __restrict__ out) {
  extern __shared__ __align__(16) char smem_raw[];
  f16* hid   = (f16*)smem_raw;          // [BM][LDH]
  f16* stage = hid + BM * LDH;          // 2 x SBUF (double-buffered tiles)

  const int g    = blockIdx.y;
  const int m0   = blockIdx.x * BM;
  const int tid  = threadIdx.x;
  const int lane = tid & 31;
  const int l16  = lane & 15;
  const int half = lane >> 4;
  const int wid  = tid >> 5;
  const int wm   = wid & 3;             // M tile (rows wm*16..)
  const int wn   = wid >> 2;            // N half (cols wn*32..)

  // staging thread mapping: x tile = 2 float4/thread; W tiles = 2 float4 each
  const int mx = tid >> 3;              // x row (and +32)
  const int qk = (tid & 7) * 4;         // x k-quad
  const int kw = tid >> 4;              // weight k row (and +16)
  const int n4 = (tid & 15) * 4;        // weight n-quad

  const float* xp = x + ((size_t)(m0 + mx) * NGRP + g) * D_IN + qk;
  const size_t xrow32  = (size_t)32 * NGRP * D_IN;
  const size_t gw_base = ((size_t)g * D_IN + kw) * D_HID + n4;
  const size_t dw_base = ((size_t)g * D_HID + kw) * D_OUT + n4;

  // ---------------- Phase A: hidden = silu(x@Wg) * (x@Wu), kept in LDS ------
  for (int hc = 0; hc < D_HID; hc += BN) {
    v8f ag0 = {}, ag1 = {}, au0 = {}, au1 = {};
    f32x4 rx0, rx1, rg0, rg1, ru0, ru1;

    // prologue: fetch + commit tile kb=0 into buffer 0
    {
      const float* gp = gate_w + gw_base + hc;
      const float* up = up_w   + gw_base + hc;
      rx0 = *(const f32x4*)(xp);
      rx1 = *(const f32x4*)(xp + xrow32);
      rg0 = *(const f32x4*)(gp);
      rg1 = *(const f32x4*)(gp + (size_t)16 * D_HID);
      ru0 = *(const f32x4*)(up);
      ru1 = *(const f32x4*)(up + (size_t)16 * D_HID);
      store_stageA(stage, stage + BM * LDK, stage + 2 * BM * LDK,
                   mx, qk, kw, n4, rx0, rx1, rg0, rg1, ru0, ru1);
    }
    __syncthreads();

    int p = 0;
    for (int kb = 0; kb < D_IN; kb += BK) {
      const bool more = (kb + BK) < D_IN;
      if (more) {  // issue next tile's global loads before the WMMAs
        const float* gp = gate_w + gw_base + (size_t)(kb + BK) * D_HID + hc;
        const float* up = up_w   + gw_base + (size_t)(kb + BK) * D_HID + hc;
        rx0 = *(const f32x4*)(xp + kb + BK);
        rx1 = *(const f32x4*)(xp + xrow32 + kb + BK);
        rg0 = *(const f32x4*)(gp);
        rg1 = *(const f32x4*)(gp + (size_t)16 * D_HID);
        ru0 = *(const f32x4*)(up);
        ru1 = *(const f32x4*)(up + (size_t)16 * D_HID);
        if (kb + 2 * BK < D_IN) {
          __builtin_prefetch(gp + (size_t)BK * D_HID, 0, 0);
          __builtin_prefetch(up + (size_t)BK * D_HID, 0, 0);
        }
      }
      {  // batch all fragment loads, then the WMMA group
        const f16* xs = stage + p * SBUF;
        const f16* wg = xs + BM * LDK;
        const f16* wu = wg + BM * LDK;
        v16h a  = frag_a(xs + (wm * 16 + l16) * LDK, half);
        v16h b0 = frag_b(wg + (wn * 32 + l16) * LDK, half);
        v16h b1 = frag_b(wg + (wn * 32 + 16 + l16) * LDK, half);
        v16h b2 = frag_b(wu + (wn * 32 + l16) * LDK, half);
        v16h b3 = frag_b(wu + (wn * 32 + 16 + l16) * LDK, half);
        ag0 = wmma16(a, b0, ag0);
        ag1 = wmma16(a, b1, ag1);
        au0 = wmma16(a, b2, au0);
        au1 = wmma16(a, b3, au1);
      }
      if (more) {  // commit next tile into the other buffer
        f16* xs = stage + (p ^ 1) * SBUF;
        store_stageA(xs, xs + BM * LDK, xs + 2 * BM * LDK,
                     mx, qk, kw, n4, rx0, rx1, rg0, rg1, ru0, ru1);
      }
      __syncthreads();
      p ^= 1;
    }

    // fused SwiGLU epilogue -> f16 hidden tile in LDS.
    // C/D layout: VGPR r holds (M = half*8 + r, N = lane%16).
#pragma unroll
    for (int r = 0; r < 8; ++r) {
      int m = wm * 16 + half * 8 + r;
      hid[m * LDH + hc + wn * 32 + l16]      = (f16)(silu(ag0[r]) * au0[r]);
      hid[m * LDH + hc + wn * 32 + 16 + l16] = (f16)(silu(ag1[r]) * au1[r]);
    }
  }

  // ---------------- Phase B: out = hidden @ Wd ------------------------------
  for (int oc = 0; oc < D_OUT; oc += BN) {
    v8f a0 = {}, a1 = {};
    f32x4 rd0, rd1;

    {  // prologue
      const float* dp = down_w + dw_base + oc;
      rd0 = *(const f32x4*)(dp);
      rd1 = *(const f32x4*)(dp + (size_t)16 * D_OUT);
      f16* wd = stage + BM * LDK;
      f16x4 d0 = cvt4(rd0), d1 = cvt4(rd1);
#pragma unroll
      for (int j = 0; j < 4; ++j) {
        wd[(n4 + j) * LDK + kw]      = d0[j];
        wd[(n4 + j) * LDK + kw + 16] = d1[j];
      }
    }
    __syncthreads();

    int p = 0;
    for (int kb = 0; kb < D_HID; kb += BK) {
      const bool more = (kb + BK) < D_HID;
      if (more) {
        const float* dp = down_w + dw_base + (size_t)(kb + BK) * D_OUT + oc;
        rd0 = *(const f32x4*)(dp);
        rd1 = *(const f32x4*)(dp + (size_t)16 * D_OUT);
        if (kb + 2 * BK < D_HID)
          __builtin_prefetch(dp + (size_t)BK * D_OUT, 0, 0);
      }
      {
        const f16* wd = stage + p * SBUF + BM * LDK;
        v16h a  = frag_a(hid + (wm * 16 + l16) * LDH + kb, half);
        v16h b0 = frag_b(wd + (wn * 32 + l16) * LDK, half);
        v16h b1 = frag_b(wd + (wn * 32 + 16 + l16) * LDK, half);
        a0 = wmma16(a, b0, a0);
        a1 = wmma16(a, b1, a1);
      }
      if (more) {
        f16* wd = stage + (p ^ 1) * SBUF + BM * LDK;
        f16x4 d0 = cvt4(rd0), d1 = cvt4(rd1);
#pragma unroll
        for (int j = 0; j < 4; ++j) {
          wd[(n4 + j) * LDK + kw]      = d0[j];
          wd[(n4 + j) * LDK + kw + 16] = d1[j];
        }
      }
      __syncthreads();
      p ^= 1;
    }

    // out is [N*G, D_OUT] with row = n*G + g
#pragma unroll
    for (int r = 0; r < 8; ++r) {
      int m = m0 + wm * 16 + half * 8 + r;
      size_t rbase = ((size_t)m * NGRP + g) * D_OUT + oc + wn * 32;
      out[rbase + l16]      = a0[r];
      out[rbase + 16 + l16] = a1[r];
    }
  }
}

extern "C" void kernel_launch(void* const* d_in, const int* in_sizes, int n_in,
                              void* d_out, int out_size, void* d_ws, size_t ws_size,
                              hipStream_t stream) {
  (void)in_sizes; (void)n_in; (void)out_size; (void)d_ws; (void)ws_size;
  const float* x  = (const float*)d_in[0];
  const float* gw = (const float*)d_in[1];
  const float* uw = (const float*)d_in[2];
  const float* dw = (const float*)d_in[3];
  float* out = (float*)d_out;

  const size_t shmem =
      (size_t)(BM * LDH + 2 * SBUF) * sizeof(_Float16);  // 293,888 B < 320 KB
  (void)hipFuncSetAttribute((const void*)grouped_swiglu_mlp,
                            hipFuncAttributeMaxDynamicSharedMemorySize,
                            (int)shmem);
  dim3 grid(N_TOK / BM, NGRP);
  grouped_swiglu_mlp<<<grid, 256, shmem, stream>>>(x, gw, uw, dw, out);
}
